// SlidingWindowAttention_75892072120387
// MI455X (gfx1250) — matrix operand order repaired and checked by reading, not compiled, not run on hardware
//
#include <hip/hip_runtime.h>

// MI455X / gfx1250, wave32. All matmuls via v_wmma_f32_16x16x32_bf16.
// B staging via GLOBAL_LOAD_ASYNC_TO_LDS_B128 (ASYNCcnt) when available.
// Shapes: B=2, T=2048, D=768, H=12, K=V=64, W=1024.

typedef __bf16 bf16;
typedef __attribute__((ext_vector_type(16))) __bf16 v16bf;
typedef __attribute__((ext_vector_type(8)))  __bf16 v8bf;
typedef __attribute__((ext_vector_type(8)))  float  v8f;
typedef __attribute__((ext_vector_type(4)))  int    v4i;

#define TT   2048
#define DD   768
#define HH   12
#define KK   64
#define WIN  1024
#define BT   4096   // B*T

#if defined(__has_builtin)
#if __has_builtin(__builtin_amdgcn_global_load_async_to_lds_b128) && \
    __has_builtin(__builtin_amdgcn_s_wait_asynccnt)
#define HAVE_ASYNC_LDS 1
#endif
#endif
#ifndef HAVE_ASYNC_LDS
#define HAVE_ASYNC_LDS 0
#endif

__device__ __forceinline__ v8f wmma_bf16(v16bf a, v16bf b, v8f c) {
  return __builtin_amdgcn_wmma_f32_16x16x32_bf16(false, a, false, b, (short)0, c,
                                                 false, false);
}

// ---- A fragment (16x32 bf16): lane holds row (lane&15); 16 contiguous
// K values: element e -> K=kc+(lane>>4)*16+e.
__device__ __forceinline__ v16bf load_a_frag(const bf16* base, int ld, int row0,
                                             int kc, int lane) {
  const bf16* p = base + (size_t)(row0 + (lane & 15)) * ld + kc + ((lane >> 4) * 16);
  v8bf lo = *(const v8bf*)(p);
  v8bf hi = *(const v8bf*)(p + 8);
  v16bf r;
#pragma unroll
  for (int i = 0; i < 8; ++i) { r[i] = lo[i]; r[i + 8] = hi[i]; }
  return r;
}

// ---- B fragment (32x16 bf16): lane holds column (lane&15), 16 contiguous
// K values starting at (lane>>4)*16. `p` must already point at them.
__device__ __forceinline__ v16bf load_b_rows(const bf16* p) {
  v8bf lo = *(const v8bf*)(p);
  v8bf hi = *(const v8bf*)(p + 8);
  v16bf r;
#pragma unroll
  for (int i = 0; i < 8; ++i) { r[i] = lo[i]; r[i + 8] = hi[i]; }
  return r;
}

// ---- 16-byte global -> LDS stage (async DMA when available) ----
__device__ __forceinline__ void stage16(const bf16* g, bf16* l) {
#if HAVE_ASYNC_LDS
  typedef __attribute__((address_space(1))) v4i gv4i;   // global int4
  typedef __attribute__((address_space(3))) v4i lv4i;   // LDS int4
  // generic->as(1): same 64-bit VA; generic->as(3): low 32 bits are LDS offset.
  __builtin_amdgcn_global_load_async_to_lds_b128(
      (gv4i*)(size_t)g, (lv4i*)(unsigned)(size_t)l, 0, 0);
#else
  *(v8bf*)l = *(const v8bf*)g;
#endif
}
__device__ __forceinline__ void wait_async_lds() {
#if HAVE_ASYNC_LDS
  __builtin_amdgcn_s_wait_asynccnt(0);
#endif
}

// =====================  RMSNorm -> bf16  =====================
__global__ __launch_bounds__(256) void rmsnorm_kernel(const float* __restrict__ x,
                                                      const float* __restrict__ w,
                                                      bf16* __restrict__ xn) {
  __shared__ float red[256];
  int row = blockIdx.x;
  const float* xr = x + (size_t)row * DD;
  float s = 0.f;
  for (int d = threadIdx.x; d < DD; d += 256) { float v = xr[d]; s += v * v; }
  red[threadIdx.x] = s;
  __syncthreads();
  for (int st = 128; st > 0; st >>= 1) {
    if (threadIdx.x < st) red[threadIdx.x] += red[threadIdx.x + st];
    __syncthreads();
  }
  float inv = rsqrtf(red[0] / (float)DD + 1e-6f);
  for (int d = threadIdx.x; d < DD; d += 256)
    xn[(size_t)row * DD + d] = (bf16)(xr[d] * inv * w[d]);
}

// =====================  Weight repack to B-fragment order  =====================
// frag(kt,nt): lane L, element e  <-  W[(kt*32+(L>>4)*16+e)*N + nt*16+(L&15)]
__global__ __launch_bounds__(256) void repack_kernel(const float* __restrict__ W,
                                                     bf16* __restrict__ out,
                                                     int Kd, int N) {
  int t = blockIdx.x * 256 + threadIdx.x;
  int total = (Kd / 32) * (N / 16) * 32;
  if (t >= total) return;
  int L = t & 31, frag = t >> 5;
  int nt = frag % (N / 16), kt = frag / (N / 16);
  int col = nt * 16 + (L & 15);
  int kb  = kt * 32 + ((L >> 4) * 16);
  bf16* dst = out + (size_t)frag * 512 + L * 16;
#pragma unroll
  for (int e = 0; e < 16; ++e) dst[e] = (bf16)W[(size_t)(kb + e) * N + col];
}

// =====================  Generic WMMA GEMM, fused epilogues  =====================
// B fragments staged once per block into LDS (double-buffered async DMA);
// all 8 waves share them instead of 8x-duplicated global loads.
// MODE 0: bf16 -> (b,h,t,k) layout, *scale        (q / k projections)
// MODE 1: relu, bf16 -> (b,h,t,k)                 (qg projection)
// MODE 2: bf16 -> V^T layout (b,h,vdim,t)         (v projection)
// MODE 3: outF[m,n] = aux[m,n] + C                (x + local)
// MODE 4: outF[m,n] += aux[m] * C                 (+ gate_mean * retrieval)
template <int MODE>
__global__ __launch_bounds__(256) void gemm_ws(const bf16* __restrict__ A,
                                               const bf16* __restrict__ Bp,
                                               int M, int N, int Kd,
                                               const float* __restrict__ aux,
                                               float* __restrict__ outF,
                                               bf16* __restrict__ outB,
                                               float scale) {
  __shared__ bf16 bstage[2][2048];           // 2 x (4 frags x 512) bf16 = 2 x 4KB
  int tid = threadIdx.x;
  int lane = tid & 31, wave = tid >> 5;
  int m0 = blockIdx.x * 128 + wave * 16;
  int n0 = blockIdx.y * 64;
  int Ndiv16 = N >> 4;
  int nk = Kd / 32;

  // per k-step: 4 consecutive fragments (nt=0..3) = 4KB contiguous in Bp
  {
    const bf16* g = Bp + ((size_t)0 * Ndiv16 + (n0 >> 4)) * 512 + tid * 8;
    stage16(g, &bstage[0][tid * 8]);
  }

  v8f acc[4] = {};
  for (int kt = 0; kt < nk; ++kt) {
    int cur = kt & 1;
    wait_async_lds();
    __syncthreads();
    if (kt + 1 < nk) {
      const bf16* g = Bp + ((size_t)(kt + 1) * Ndiv16 + (n0 >> 4)) * 512 + tid * 8;
      stage16(g, &bstage[cur ^ 1][tid * 8]);
    }
    v16bf a = load_a_frag(A, Kd, m0, kt * 32, lane);
#pragma unroll
    for (int nt = 0; nt < 4; ++nt) {
      v16bf b = *(const v16bf*)&bstage[cur][nt * 512 + lane * 16];
      acc[nt] = wmma_bf16(a, b, acc[nt]);
    }
  }

  int mrow = m0 + ((lane >> 4) * 8);
#pragma unroll
  for (int nt = 0; nt < 4; ++nt) {
#pragma unroll
    for (int i = 0; i < 8; ++i) {
      int m = mrow + i;
      int n = n0 + nt * 16 + (lane & 15);
      float c = acc[nt][i];
      if (MODE == 0) {
        int b = m >> 11, t = m & (TT - 1), h = n >> 6, kk = n & 63;
        outB[((size_t)(b * HH + h) * TT + t) * KK + kk] = (bf16)(c * scale);
      } else if (MODE == 1) {
        int b = m >> 11, t = m & (TT - 1), h = n >> 6, kk = n & 63;
        outB[((size_t)(b * HH + h) * TT + t) * KK + kk] = (bf16)(c > 0.f ? c : 0.f);
      } else if (MODE == 2) {
        int b = m >> 11, t = m & (TT - 1), h = n >> 6, vd = n & 63;
        outB[((size_t)(b * HH + h) * KK + vd) * TT + t] = (bf16)c;
      } else if (MODE == 3) {
        outF[(size_t)m * N + n] = aux[(size_t)m * N + n] + c;
      } else {
        outF[(size_t)m * N + n] += aux[m] * c;
      }
    }
  }
}

// =====================  Sliding-window flash attention  =====================
// One wave per 16-query tile; 32-key tiles; online softmax; P via per-wave LDS.
// Mask logic only on boundary tiles (wave-uniform branch); interior tiles run
// the unmasked fast path.
__global__ __launch_bounds__(256) void attn_kernel(const bf16* __restrict__ qbf,
                                                   const bf16* __restrict__ kbf,
                                                   const bf16* __restrict__ vT,
                                                   bf16* __restrict__ Obf) {
  __shared__ bf16 plds[8][16 * 40];   // stride 40 -> rows 80B apart (16B-aligned)
  int lane = threadIdx.x & 31, wave = threadIdx.x >> 5;
  int bh = blockIdx.x >> 4;                       // 16 blocks of 128 queries per (b,h)
  int q0 = (blockIdx.x & 15) * 128 + wave * 16;
  int b = bh / HH, h = bh % HH;
  const bf16* qb = qbf + (size_t)bh * TT * KK;
  const bf16* kb = kbf + (size_t)bh * TT * KK;
  const bf16* vb = vT  + (size_t)bh * KK * TT;

  v16bf aq0 = load_a_frag(qb, KK, q0, 0, lane);
  v16bf aq1 = load_a_frag(qb, KK, q0, 32, lane);

  float mi[8], li[8];
  v8f oacc[4] = {};
#pragma unroll
  for (int i = 0; i < 8; ++i) { mi[i] = -1e30f; li[i] = 0.f; }

  int lo = q0 - WIN; if (lo < 0) lo = 0; lo &= ~31;
  int rbase = q0 + ((lane >> 4) * 8);
  bf16* pw = &plds[wave][0];

  for (int s0 = lo; s0 <= q0 + 15; s0 += 32) {
    bool needmask = (s0 + 31 > q0) || (s0 < q0 + 15 - WIN);   // wave-uniform
    // prefetch next key tile while this one computes
    int sn = s0 + 32;
    if (sn <= q0 + 15) {
      __builtin_prefetch((const void*)(kb + (size_t)(sn + lane) * KK), 0, 0);
      __builtin_prefetch((const void*)(vb + (size_t)(lane * 2) * TT + sn), 0, 0);
    }
    // ---- S = Q K^T (two 16-key halves, K=64 in two 32-slices) ----
    v8f s[2] = {};
#pragma unroll
    for (int n16 = 0; n16 < 2; ++n16) {
      const bf16* kp = kb + (size_t)(s0 + n16 * 16 + (lane & 15)) * KK + ((lane >> 4) * 16);
      v16bf bk0 = load_b_rows(kp);
      v16bf bk1 = load_b_rows(kp + 32);
      s[n16] = wmma_bf16(aq0, bk0, s[n16]);
      s[n16] = wmma_bf16(aq1, bk1, s[n16]);
    }
    if (needmask) {
#pragma unroll
      for (int i = 0; i < 8; ++i) {
        int row = rbase + i;
#pragma unroll
        for (int n16 = 0; n16 < 2; ++n16) {
          int col = s0 + n16 * 16 + (lane & 15);
          float v = s[n16][i];
          if (col > row || col + WIN < row) v = -1e30f;
          s[n16][i] = v;
        }
      }
    }
    // ---- online softmax (row = 16 lanes of a half-wave) ----
    float alpha[8];
#pragma unroll
    for (int i = 0; i < 8; ++i) {
      float rm = fmaxf(s[0][i], s[1][i]);
      for (int off = 1; off < 16; off <<= 1) rm = fmaxf(rm, __shfl_xor(rm, off, 32));
      float mnew = fmaxf(mi[i], rm);
      alpha[i] = __expf(mi[i] - mnew);
      float p0, p1;
      if (needmask) {
        p0 = (s[0][i] < -9e29f) ? 0.f : __expf(s[0][i] - mnew);
        p1 = (s[1][i] < -9e29f) ? 0.f : __expf(s[1][i] - mnew);
      } else {
        p0 = __expf(s[0][i] - mnew);
        p1 = __expf(s[1][i] - mnew);
      }
      s[0][i] = p0; s[1][i] = p1;
      float rs = p0 + p1;
      for (int off = 1; off < 16; off <<= 1) rs += __shfl_xor(rs, off, 32);
      li[i] = li[i] * alpha[i] + rs;
      mi[i] = mnew;
    }
    // ---- P: C-layout -> LDS -> A-layout (DS ops in-order within wave) ----
#pragma unroll
    for (int i = 0; i < 8; ++i) {
      int M = i + ((lane >> 4) * 8);
      pw[M * 40 + (lane & 15)]      = (bf16)s[0][i];
      pw[M * 40 + 16 + (lane & 15)] = (bf16)s[1][i];
    }
    v16bf pa = load_a_frag(pw, 40, 0, 0, lane);
    // ---- O = O*alpha + P V ----
#pragma unroll
    for (int nv = 0; nv < 4; ++nv)
#pragma unroll
      for (int i = 0; i < 8; ++i) oacc[nv][i] *= alpha[i];
#pragma unroll
    for (int nv = 0; nv < 4; ++nv) {
      const bf16* vp = vb + (size_t)(nv * 16 + (lane & 15)) * TT + s0 + ((lane >> 4) * 16);
      v16bf bv = load_b_rows(vp);
      oacc[nv] = wmma_bf16(pa, bv, oacc[nv]);
    }
  }
  // ---- normalize + store (b,t,h*64) bf16 ----
#pragma unroll
  for (int nv = 0; nv < 4; ++nv)
#pragma unroll
    for (int i = 0; i < 8; ++i) {
      float o = oacc[nv][i] / li[i];
      Obf[(size_t)(b * TT + rbase + i) * DD + h * KK + nv * 16 + (lane & 15)] = (bf16)o;
    }
}

// =====================  GDN retrieval: relu(qg) @ state  =====================
__global__ __launch_bounds__(256) void retrieval_kernel(const bf16* __restrict__ qg,
                                                        const float* __restrict__ state,
                                                        bf16* __restrict__ R) {
  __shared__ bf16 stT[64 * 64];   // state^T: stT[vdim][kk]
  int bh = blockIdx.x, b = bh / HH, h = bh % HH;
  const float* st = state + (size_t)bh * 64 * 64;
  for (int idx = threadIdx.x; idx < 4096; idx += 256) {
    int kk = idx >> 6, vd = idx & 63;
    stT[vd * 64 + kk] = (bf16)st[idx];
  }
  __syncthreads();
  int lane = threadIdx.x & 31, wave = threadIdx.x >> 5;
  v16bf bs[2][4];
#pragma unroll
  for (int c = 0; c < 2; ++c)
#pragma unroll
    for (int nt = 0; nt < 4; ++nt)
      bs[c][nt] = *(const v16bf*)&stT[(nt * 16 + (lane & 15)) * 64 + c * 32 + ((lane >> 4) * 16)];
  const bf16* qgb = qg + (size_t)bh * TT * KK;
  for (int t0 = wave * 16; t0 < TT; t0 += 128) {
    v16bf a0 = load_a_frag(qgb, KK, t0, 0, lane);
    v16bf a1 = load_a_frag(qgb, KK, t0, 32, lane);
    v8f acc[4] = {};
#pragma unroll
    for (int nt = 0; nt < 4; ++nt) {
      acc[nt] = wmma_bf16(a0, bs[0][nt], acc[nt]);
      acc[nt] = wmma_bf16(a1, bs[1][nt], acc[nt]);
    }
    int rbase = t0 + ((lane >> 4) * 8);
#pragma unroll
    for (int nt = 0; nt < 4; ++nt)
#pragma unroll
      for (int i = 0; i < 8; ++i)
        R[(size_t)(b * TT + rbase + i) * DD + h * KK + nt * 16 + (lane & 15)] = (bf16)acc[nt][i];
  }
}

// =====================  gate mean  =====================
__global__ __launch_bounds__(256) void gate_kernel(const bf16* __restrict__ xn,
                                                   const float* __restrict__ Wg,
                                                   const float* __restrict__ bg,
                                                   float* __restrict__ gmean) {
  int id = blockIdx.x * 256 + threadIdx.x;
  if (id >= BT) return;
  float acc[HH];
#pragma unroll
  for (int h = 0; h < HH; ++h) acc[h] = bg[h];
  const bf16* xr = xn + (size_t)id * DD;
  for (int d = 0; d < DD; ++d) {
    float xv = (float)xr[d];
    const float* wr = Wg + (size_t)d * HH;
#pragma unroll
    for (int h = 0; h < HH; ++h) acc[h] += xv * wr[h];
  }
  float s = 0.f;
#pragma unroll
  for (int h = 0; h < HH; ++h) s += 1.f / (1.f + __expf(-acc[h]));
  gmean[id] = s / (float)HH;
}

// =====================  host orchestration  =====================
extern "C" void kernel_launch(void* const* d_in, const int* in_sizes, int n_in,
                              void* d_out, int out_size, void* d_ws, size_t ws_size,
                              hipStream_t stream) {
  const float* x    = (const float*)d_in[0];
  const float* gdn  = (const float*)d_in[1];
  const float* Wq   = (const float*)d_in[2];
  const float* Wk   = (const float*)d_in[3];
  const float* Wv   = (const float*)d_in[4];
  const float* Wo   = (const float*)d_in[5];
  const float* Wgq  = (const float*)d_in[6];
  const float* Wro  = (const float*)d_in[7];
  const float* Wg   = (const float*)d_in[8];
  const float* bg   = (const float*)d_in[9];
  const float* nw   = (const float*)d_in[10];
  float* out = (float*)d_out;

  char* ws = (char*)d_ws;
  size_t off = 0;
  auto take = [&](size_t bytes) {
    size_t r = off; off += (bytes + 255) & ~(size_t)255; return r;
  };
  const size_t ROWB = (size_t)BT * DD * sizeof(bf16);   // 6.29 MB
  const size_t WPB  = (size_t)DD * DD * sizeof(bf16);   // 1.18 MB
  bf16* xn   = (bf16*)(ws + take(ROWB));
  bf16* wqp  = (bf16*)(ws + take(WPB));
  bf16* wkp  = (bf16*)(ws + take(WPB));
  bf16* wvp  = (bf16*)(ws + take(WPB));
  bf16* wgqp = (bf16*)(ws + take(WPB));
  bf16* wop  = (bf16*)(ws + take(WPB));
  bf16* wrop = (bf16*)(ws + take(WPB));
  bf16* qbf  = (bf16*)(ws + take(ROWB));
  bf16* kbf  = (bf16*)(ws + take(ROWB));
  bf16* vTb  = (bf16*)(ws + take(ROWB));
  bf16* qgbf = (bf16*)(ws + take(ROWB));
  bf16* Obf  = (bf16*)(ws + take(ROWB));
  bf16* Rbf  = (bf16*)(ws + take(ROWB));
  float* gm  = (float*)(ws + take((size_t)BT * sizeof(float)));
  (void)ws_size; (void)in_sizes; (void)n_in; (void)out_size;

  rmsnorm_kernel<<<BT, 256, 0, stream>>>(x, nw, xn);

  int rp_blocks = ((DD / 32) * (DD / 16) * 32 + 255) / 256;
  repack_kernel<<<rp_blocks, 256, 0, stream>>>(Wq,  wqp,  DD, DD);
  repack_kernel<<<rp_blocks, 256, 0, stream>>>(Wk,  wkp,  DD, DD);
  repack_kernel<<<rp_blocks, 256, 0, stream>>>(Wv,  wvp,  DD, DD);
  repack_kernel<<<rp_blocks, 256, 0, stream>>>(Wgq, wgqp, DD, DD);
  repack_kernel<<<rp_blocks, 256, 0, stream>>>(Wo,  wop,  DD, DD);
  repack_kernel<<<rp_blocks, 256, 0, stream>>>(Wro, wrop, DD, DD);

  dim3 gg(BT / 128, DD / 64);
  gemm_ws<0><<<gg, 256, 0, stream>>>(xn, wqp,  BT, DD, DD, nullptr, nullptr, qbf, 0.125f);
  gemm_ws<0><<<gg, 256, 0, stream>>>(xn, wkp,  BT, DD, DD, nullptr, nullptr, kbf, 1.0f);
  gemm_ws<2><<<gg, 256, 0, stream>>>(xn, wvp,  BT, DD, DD, nullptr, nullptr, vTb, 1.0f);
  gemm_ws<1><<<gg, 256, 0, stream>>>(xn, wgqp, BT, DD, DD, nullptr, nullptr, qgbf, 1.0f);

  attn_kernel<<<2 * HH * (TT / 128), 256, 0, stream>>>(qbf, kbf, vTb, Obf);
  retrieval_kernel<<<2 * HH, 256, 0, stream>>>(qgbf, gdn, Rbf);
  gate_kernel<<<(BT + 255) / 256, 256, 0, stream>>>(xn, Wg, bg, gm);

  gemm_ws<3><<<gg, 256, 0, stream>>>(Obf, wop,  BT, DD, DD, x,  out, nullptr, 1.0f);
  gemm_ws<4><<<gg, 256, 0, stream>>>(Rbf, wrop, BT, DD, DD, gm, out, nullptr, 1.0f);
}
